// LaneGNN_36318243455493
// MI455X (gfx1250) — matrix operand order repaired and checked by reading, not compiled, not run on hardware
//
#include <hip/hip_runtime.h>
#include <math.h>
#include <stdint.h>

// ---------------------------------------------------------------------------
// LaneGNN fused pipeline for MI455X (gfx1250, wave32).
//
//   B=4, N=512, FD=256, FI=128, IOU=1.
//
// Algebraic collapse: the 34.4-GFLOP edge GEMM  (e @ We1^T over B*N*N rows)
// distributes over the additive structure of e, so
//   y[b,i,j,n] = Yin[b,i,n] + Yb[b,j,n] + pe0*u0[n] + pe1*u1[n]
// with Yin = nf @ (We1 Wi)^T, Yb = cb - nf @ (We1 Wo)^T   (tiny WMMA GEMMs),
// u0/u1 = We1 Wp[:,k], cb = We1 (bi-bo+bp) + be1.
// The fused edge kernel applies exact GELU, the We2 dot, the supress mask and
// the running max over i without ever materializing the 537 MB [B,N,N,128]
// tensor.  Remaining HBM traffic ~25 MB (~1.1 us @ 23.3 TB/s); the kernel is
// VALU/transcendental bound (~1.2G ops + 134M erf).
//
// Round-1 feedback fix: the WMMA GEMM previously held 8 A fragments live
// across n-chunks -> scratch spills (scratch_load_b128 in the inner loop).
// The loop nest is now  nc { stage B; kt { build 1 A frag; nt { wmma } } }
// so live VGPRs = 8 (A) + 32 (acc) + 8 (B) and nothing spills.
// ---------------------------------------------------------------------------

typedef __attribute__((ext_vector_type(16))) _Float16 v16h;
typedef __attribute__((ext_vector_type(8)))  float    v8f;

#define B_   4
#define N_   512
#define FD_  256
#define FI_  128
#define M2_  (B_ * N_)      // 2048 rows for the batched GEMMs

__device__ __forceinline__ float gelu_exact(float x) {
    // exact GELU, matches jax.nn.gelu(approximate=False)
    return 0.5f * x * (1.0f + erff(x * 0.70710678118654752f));
}

// ---------------------------------------------------------------------------
// P0: fold the linear layers:  Ci = We1*Wi, CoNeg = -(We1*Wo),
//     cb = We1*(bi-bo+bp)+be1,  u4[n] = (We1*Wp[:,0], We1*Wp[:,1], We2[n], 0)
// ---------------------------------------------------------------------------
__global__ __launch_bounds__(256)
void lanegnn_fold(const float* __restrict__ Wi, const float* __restrict__ Wo,
                  const float* __restrict__ Wp, const float* __restrict__ bi,
                  const float* __restrict__ bo, const float* __restrict__ bp,
                  const float* __restrict__ We1, const float* __restrict__ be1,
                  const float* __restrict__ We2,
                  float* __restrict__ Ci, float* __restrict__ CoNeg,
                  float* __restrict__ cb, float4* __restrict__ u4) {
    int blk = blockIdx.x;
    int t   = threadIdx.x;
    if (blk < FI_) {                       // Ci row g = blk, col d = t (0..255)
        int g = blk;
        float s = 0.f;
        #pragma unroll 4
        for (int f = 0; f < FI_; ++f) s = fmaf(We1[g * FI_ + f], Wi[f * FD_ + t], s);
        Ci[g * FD_ + t] = s;
    } else if (blk < 2 * FI_) {            // CoNeg
        int g = blk - FI_;
        float s = 0.f;
        #pragma unroll 4
        for (int f = 0; f < FI_; ++f) s = fmaf(We1[g * FI_ + f], Wo[f * FD_ + t], s);
        CoNeg[g * FD_ + t] = -s;
    } else {                               // vectors
        if (t < FI_) {
            int g = t;
            float s0 = 0.f, s1 = 0.f, sc = 0.f;
            for (int f = 0; f < FI_; ++f) {
                float w = We1[g * FI_ + f];
                s0 = fmaf(w, Wp[f * 2 + 0], s0);
                s1 = fmaf(w, Wp[f * 2 + 1], s1);
                sc = fmaf(w, bi[f] - bo[f] + bp[f], sc);
            }
            cb[g] = sc + be1[g];
            u4[g] = make_float4(s0, s1, We2[g], 0.f);
        }
    }
}

// ---------------------------------------------------------------------------
// WMMA GEMM:  out[M2_, NOUT] = act(A[M2_,256] @ W[NOUT,256]^T + bias)
// f32 inputs converted to f16 fragments, f32 accumulation via
// v_wmma_f32_16x16x32_f16.  8 waves/block, 16 rows/wave, B staged in LDS in
// a contiguous-K 32x16 fragment layout (16 consecutive K per lane).
// ---------------------------------------------------------------------------
template <int NOUT, bool RELU, bool HASBIAS>
__global__ __launch_bounds__(256)
void lanegnn_gemm(const float* __restrict__ A, const float* __restrict__ W,
                  const float* __restrict__ bias, float* __restrict__ out) {
    constexpr int K   = 256;
    constexpr int NKT = K / 32;     // 8 k-tiles
    constexpr int NCH = 64;         // output columns per LDS chunk
    constexpr int NTC = NCH / 16;   // 4 n-tiles per chunk

    __shared__ v16h bfr[NKT][NTC][32];   // 32 KB of f16 B fragments

    const int tid = threadIdx.x;
    const int w   = tid >> 5;
    const int ln  = tid & 31;
    const int m   = ln & 15;        // row within 16x16 tile / C column index
    const int kh  = ln >> 4;        // K-half selector (A/B fragment layout)
    const int rowBase = blockIdx.x * 128 + w * 16;
    const int row     = rowBase + m;
    const float* arow = A + (size_t)row * K;

    for (int nc = 0; nc < NOUT / NCH; ++nc) {
        __syncthreads();
        // Stage B fragments: 8*4*32 = 1024 lane-slots, 4 per thread.
        #pragma unroll
        for (int s = 0; s < 4; ++s) {
            int id = tid + 256 * s;
            int kt = id >> 7, nt = (id >> 5) & 3, l2 = id & 31;
            int n  = nc * NCH + nt * 16 + (l2 & 15);
            int k0 = 32 * kt + 16 * (l2 >> 4);       // 16 consecutive K
            const float4* wp = (const float4*)(W + (size_t)n * K + k0);
            float4 q0 = wp[0], q1 = wp[1], q2 = wp[2], q3 = wp[3];
            v16h bv;
            bv[0]  = (_Float16)q0.x; bv[1]  = (_Float16)q0.y;
            bv[2]  = (_Float16)q0.z; bv[3]  = (_Float16)q0.w;
            bv[4]  = (_Float16)q1.x; bv[5]  = (_Float16)q1.y;
            bv[6]  = (_Float16)q1.z; bv[7]  = (_Float16)q1.w;
            bv[8]  = (_Float16)q2.x; bv[9]  = (_Float16)q2.y;
            bv[10] = (_Float16)q2.z; bv[11] = (_Float16)q2.w;
            bv[12] = (_Float16)q3.x; bv[13] = (_Float16)q3.y;
            bv[14] = (_Float16)q3.z; bv[15] = (_Float16)q3.w;
            bfr[kt][nt][l2] = bv;
        }
        __syncthreads();

        // Accumulators for the 4 n-tiles of this chunk (32 VGPRs).
        v8f acc0 = {}, acc1 = {}, acc2 = {}, acc3 = {};
        #pragma unroll
        for (int kt = 0; kt < NKT; ++kt) {
            // Build one A fragment (16x32 f16), contiguous-K A layout:
            //   K = 16h..16h+15 (h = kh), matching the B staging
            const float4* p0 = (const float4*)(arow + 32 * kt + 16 * kh);
            const float4* p1 = (const float4*)(arow + 32 * kt + 16 * kh + 8);
            float4 q0 = p0[0], q1 = p0[1], q2 = p1[0], q3 = p1[1];
            v16h afr;
            afr[0]  = (_Float16)q0.x; afr[1]  = (_Float16)q0.y;
            afr[2]  = (_Float16)q0.z; afr[3]  = (_Float16)q0.w;
            afr[4]  = (_Float16)q1.x; afr[5]  = (_Float16)q1.y;
            afr[6]  = (_Float16)q1.z; afr[7]  = (_Float16)q1.w;
            afr[8]  = (_Float16)q2.x; afr[9]  = (_Float16)q2.y;
            afr[10] = (_Float16)q2.z; afr[11] = (_Float16)q2.w;
            afr[12] = (_Float16)q3.x; afr[13] = (_Float16)q3.y;
            afr[14] = (_Float16)q3.z; afr[15] = (_Float16)q3.w;

            acc0 = __builtin_amdgcn_wmma_f32_16x16x32_f16(
                false, afr, false, bfr[kt][0][ln], (short)0, acc0, false, false);
            acc1 = __builtin_amdgcn_wmma_f32_16x16x32_f16(
                false, afr, false, bfr[kt][1][ln], (short)0, acc1, false, false);
            acc2 = __builtin_amdgcn_wmma_f32_16x16x32_f16(
                false, afr, false, bfr[kt][2][ln], (short)0, acc2, false, false);
            acc3 = __builtin_amdgcn_wmma_f32_16x16x32_f16(
                false, afr, false, bfr[kt][3][ln], (short)0, acc3, false, false);
        }

        // Epilogue: bias + activation + store (C: col = lane%16, rows r+8*kh).
        #pragma unroll
        for (int nt = 0; nt < NTC; ++nt) {
            v8f acc = (nt == 0) ? acc0 : (nt == 1) ? acc1 : (nt == 2) ? acc2 : acc3;
            int n = nc * NCH + nt * 16 + m;
            float bv = HASBIAS ? bias[n] : 0.f;
            #pragma unroll
            for (int r = 0; r < 8; ++r) {
                int orow = rowBase + r + 8 * kh;
                float v = acc[r] + bv;
                if (RELU) v = fmaxf(v, 0.f);
                out[(size_t)orow * NOUT + n] = v;
            }
        }
    }
}

// ---------------------------------------------------------------------------
// K2: fused edge MLP + GELU + We2 dot + mask + max over i.
// One wave per output column j; 8 waves/block.  Yin tile staged per i-chunk
// in LDS (padded rows -> conflict-free column reads); Yb per-wave in LDS;
// u0/u1/We2 packed as float4 broadcast reads.
// ---------------------------------------------------------------------------
__global__ __launch_bounds__(256)
void lanegnn_edge(const float* __restrict__ Yin, const float* __restrict__ Yb,
                  const float4* __restrict__ u4, const float* __restrict__ pos,
                  const float* __restrict__ sup, const float* __restrict__ be2p,
                  float* __restrict__ nm) {
    __shared__ float4 u4S[FI_];            // (u0, u1, we2, 0) per feature
    __shared__ float  YbS[8][FI_];         // per-wave j row
    __shared__ float  YinS[16][FI_ + 4];   // padded: stride 132 kills bank conflicts

    const int tid  = threadIdx.x;
    const int w    = tid >> 5;
    const int ln   = tid & 31;
    const int b    = blockIdx.x >> 6;            // N/8 = 64 j-groups per batch
    const int j    = (blockIdx.x & 63) * 8 + w;  // one j per wave
    const int m    = ln & 15;                    // i-row within chunk
    const int half = ln >> 4;                    // feature half (0:n<64, 1:n>=64)

    if (tid < FI_) u4S[tid] = u4[tid];
    {
        const float* yb = Yb + (size_t)(b * N_ + j) * FI_;
        #pragma unroll
        for (int t = 0; t < 4; ++t) YbS[w][ln + 32 * t] = yb[ln + 32 * t];
    }
    const float be2 = be2p[0];
    float runmax = -__builtin_inff();

    for (int ib = 0; ib < N_; ib += 16) {
        __syncthreads();
        {   // cooperative, coalesced stage of the 16x128 Yin tile
            const float* yin = Yin + (size_t)(b * N_ + ib) * FI_;
            #pragma unroll
            for (int t = 0; t < 8; ++t) {
                int idx = tid + 256 * t;                 // 0..2047
                YinS[idx >> 7][idx & 127] = yin[idx];
            }
        }
        __syncthreads();

        const int i = ib + m;
        const float2 pe  = *(const float2*)(pos + (((size_t)(b * N_ + i)) * N_ + j) * 2);
        const float mask = sup[((size_t)(b * N_ + i)) * N_ + j];

        float partial = 0.f;
        const int nb = half * 64;
        #pragma unroll 4
        for (int t = 0; t < 16; ++t) {
            int n = nb + 4 * t;
            float4 yi = *(const float4*)&YinS[m][n];
            float4 yb = *(const float4*)&YbS[w][n];
            float4 u0 = u4S[n + 0], u1 = u4S[n + 1], u2 = u4S[n + 2], u3 = u4S[n + 3];
            float y;
            y = yi.x + yb.x + pe.x * u0.x + pe.y * u0.y;
            partial = fmaf(gelu_exact(y), u0.z, partial);
            y = yi.y + yb.y + pe.x * u1.x + pe.y * u1.y;
            partial = fmaf(gelu_exact(y), u1.z, partial);
            y = yi.z + yb.z + pe.x * u2.x + pe.y * u2.y;
            partial = fmaf(gelu_exact(y), u2.z, partial);
            y = yi.w + yb.w + pe.x * u3.x + pe.y * u3.y;
            partial = fmaf(gelu_exact(y), u3.z, partial);
        }
        // merge the two feature-halves of row m (wave32 xor-16)
        partial += __shfl_xor(partial, 16);
        float e = (partial + be2) * mask;
        // max over the 16 i-rows of this chunk
        #pragma unroll
        for (int off = 1; off < 16; off <<= 1) e = fmaxf(e, __shfl_xor(e, off));
        runmax = fmaxf(runmax, e);
    }
    if (ln == 0) nm[b * N_ + j] = runmax;
}

// ---------------------------------------------------------------------------
// K3a: h1[row,d] = relu(nm[row]*Wn1[d] + bn1[d])   (rank-1 + bias, trivial)
// ---------------------------------------------------------------------------
__global__ __launch_bounds__(256)
void lanegnn_h1(const float* __restrict__ nm, const float* __restrict__ Wn1,
                const float* __restrict__ bn1, float* __restrict__ h1) {
    int row = blockIdx.x, d = threadIdx.x;
    float v = fmaf(nm[row], Wn1[d], bn1[d]);
    h1[(size_t)row * FD_ + d] = fmaxf(v, 0.f);
}

// ---------------------------------------------------------------------------
extern "C" void kernel_launch(void* const* d_in, const int* in_sizes, int n_in,
                              void* d_out, int out_size, void* d_ws, size_t ws_size,
                              hipStream_t stream) {
    (void)in_sizes; (void)n_in; (void)out_size; (void)ws_size;
    const float* nf  = (const float*)d_in[0];
    const float* sup = (const float*)d_in[1];
    const float* pos = (const float*)d_in[2];
    const float* Wp  = (const float*)d_in[3];
    const float* bp  = (const float*)d_in[4];
    const float* Wi  = (const float*)d_in[5];
    const float* bi  = (const float*)d_in[6];
    const float* Wo  = (const float*)d_in[7];
    const float* bo  = (const float*)d_in[8];
    const float* We1 = (const float*)d_in[9];
    const float* be1 = (const float*)d_in[10];
    const float* We2 = (const float*)d_in[11];
    const float* be2 = (const float*)d_in[12];
    const float* Wn1 = (const float*)d_in[13];
    const float* bn1 = (const float*)d_in[14];
    const float* Wn2 = (const float*)d_in[15];
    const float* bn2 = (const float*)d_in[16];

    // workspace carve-up (~4.4 MB total)
    char* ws = (char*)d_ws;
    float*  Ci    = (float*)ws;  ws += FI_ * FD_ * 4;          // 128 KB
    float*  CoNeg = (float*)ws;  ws += FI_ * FD_ * 4;          // 128 KB
    float*  cb    = (float*)ws;  ws += FI_ * 4;
    ws = (char*)(((uintptr_t)ws + 15) & ~(uintptr_t)15);
    float4* u4    = (float4*)ws; ws += FI_ * 16;               // 2 KB
    float*  Yin   = (float*)ws;  ws += (size_t)M2_ * FI_ * 4;  // 1 MB
    float*  Yb    = (float*)ws;  ws += (size_t)M2_ * FI_ * 4;  // 1 MB
    float*  nm    = (float*)ws;  ws += M2_ * 4;                // 8 KB
    float*  h1    = (float*)ws;  ws += (size_t)M2_ * FD_ * 4;  // 2 MB

    // P0: fold linear layers into Ci/CoNeg/cb/u4
    lanegnn_fold<<<2 * FI_ + 1, 256, 0, stream>>>(Wi, Wo, Wp, bi, bo, bp,
                                                  We1, be1, We2, Ci, CoNeg, cb, u4);
    // P1: Yin = nf @ Ci^T ; Yb = nf @ CoNeg^T + cb   (WMMA)
    lanegnn_gemm<FI_, false, false><<<M2_ / 128, 256, 0, stream>>>(nf, Ci, nullptr, Yin);
    lanegnn_gemm<FI_, false, true ><<<M2_ / 128, 256, 0, stream>>>(nf, CoNeg, cb, Yb);
    // K2: fused edge MLP + GELU + We2 dot + mask + max over i
    lanegnn_edge<<<B_ * (N_ / 8), 256, 0, stream>>>(Yin, Yb, u4, pos, sup, be2, nm);
    // K3: node layers
    lanegnn_h1<<<M2_, FD_, 0, stream>>>(nm, Wn1, bn1, h1);
    lanegnn_gemm<FD_, true, true ><<<M2_ / 128, 256, 0, stream>>>(h1, Wn2, bn2, (float*)d_out);
}
